// GCNClassifier_58720792871581
// MI455X (gfx1250) — matrix operand order repaired and checked by reading, not compiled, not run on hardware
//
#include <hip/hip_runtime.h>
#include <hip/hip_bf16.h>

typedef __attribute__((ext_vector_type(16))) __bf16 v16bf;
typedef __attribute__((ext_vector_type(8)))  __bf16 v8bf;
typedef __attribute__((ext_vector_type(4)))  __bf16 v4bf;
typedef __attribute__((ext_vector_type(8)))  float  v8f;

#define GCN_N   100000
#define GCN_FIN 128
#define TB      256            // threads per block everywhere

// round-to-nearest-even f32 -> bf16 (pure integer VALU, no codegen reliance)
static __device__ __forceinline__ __bf16 f2bf(float f) {
    unsigned u = __builtin_bit_cast(unsigned, f);
    unsigned r = (u + 0x7FFFu + ((u >> 16) & 1u)) >> 16;
    unsigned short s = (unsigned short)r;
    return __builtin_bit_cast(__bf16, s);
}

__global__ void zero_f32x4(float4* __restrict__ p, long long n4) {
    long long i = (long long)blockIdx.x * blockDim.x + threadIdx.x;
    if (i < n4) p[i] = make_float4(0.f, 0.f, 0.f, 0.f);
}

__global__ void deg_count(float* __restrict__ deg, const int* __restrict__ dst, int E) {
    int e = blockIdx.x * blockDim.x + threadIdx.x;
    if (e < E) atomicAdd(&deg[dst[e]], 1.0f);
}

__global__ void deg_rsqrt(float* __restrict__ deg, int n) {
    int i = blockIdx.x * blockDim.x + threadIdx.x;
    if (i < n) deg[i] = rsqrtf(deg[i] + 1.0f);
}

// Wt[n][k] = bf16(W[k][n]); W is [128][Fout] f32, Wt is [Fout][128] bf16 (32 KB max).
__global__ void w_bf16_transpose(__bf16* __restrict__ Wt, const float* __restrict__ W,
                                 int total, int logFout) {
    int gid = blockIdx.x * blockDim.x + threadIdx.x;
    if (gid >= total) return;
    int k = gid >> logFout;
    int n = gid & ((1 << logFout) - 1);
    Wt[(n << 7) + k] = f2bf(W[gid]);
}

// out[nrows,FOUT] = X[nrows,128] @ W[128,FOUT] via v_wmma_f32_16x16x32_bf16.
// Block = 256 thr = 8 waves; block owns a 16-row tile (A staged in LDS as bf16),
// wave w owns column tile n0 = 16*w (FOUT compile-time -> immediate store offsets,
// no column loop). A/B fragments preloaded so WMMAs issue under partial waits.
template <int FOUT>
__global__ void __launch_bounds__(TB)
gemm_x128_bf16_wmma(float* __restrict__ out, const float* __restrict__ X,
                    const __bf16* __restrict__ Wt, int nrows) {
    // row stride 136 halves = 272 B (16B-aligned, +4 banks/row -> conflict-free)
    __shared__ __bf16 As[16 * 136];

    int row0 = blockIdx.x << 4;
    // stage A tile: 16x128 f32 -> bf16; float4 in, v4bf (ds_store_b64) out
#pragma unroll
    for (int c = threadIdx.x; c < 16 * 32; c += TB) {
        int r = c >> 5, c4 = (c & 31) << 2;
        int row = row0 + r;
        float4 v = (row < nrows) ? *(const float4*)(X + (long long)row * 128 + c4)
                                 : make_float4(0.f, 0.f, 0.f, 0.f);
        v4bf bv; bv[0] = f2bf(v.x); bv[1] = f2bf(v.y); bv[2] = f2bf(v.z); bv[3] = f2bf(v.w);
        *(v4bf*)(&As[r * 136 + c4]) = bv;
    }
    __syncthreads();

    int wave = threadIdx.x >> 5;
    int lane = threadIdx.x & 31;
    int half = lane >> 4;      // 0: lanes 0-15, 1: lanes 16-31
    int m    = lane & 15;      // A row within tile
    int nl   = lane & 15;      // B/C column within tile

    int n0 = wave << 4;
    if (n0 >= FOUT) return;    // uniform per wave; barrier already passed

    // A 16x32 bf16 fragment: elem i -> K = k0+16*half+i (matches B)
    // = one contiguous 16-run -> 2 x ds_load_b128 + shuffle
    auto loadA = [&](int k0) -> v16bf {
        const __bf16* ap = &As[m * 136 + k0 + (half << 4)];
        v8bf lo = *(const v8bf*)ap;
        v8bf hi = *(const v8bf*)(ap + 8);
        return __builtin_shufflevector(lo, hi, 0, 1, 2, 3, 4, 5, 6, 7,
                                       8, 9, 10, 11, 12, 13, 14, 15);
    };
    v16bf a0 = loadA(0), a1 = loadA(32), a2 = loadA(64), a3 = loadA(96);

    // B 32x16 fragment: elem i -> K = k0+16*half+i (contiguous in this lane's Wt row)
    const __bf16* wrow = Wt + ((long long)(n0 + nl) << 7) + (half << 4);
    v16bf b0 = *(const v16bf*)(wrow + 0);
    v16bf b1 = *(const v16bf*)(wrow + 32);
    v16bf b2 = *(const v16bf*)(wrow + 64);
    v16bf b3 = *(const v16bf*)(wrow + 96);

    v8f acc = {};
    acc = __builtin_amdgcn_wmma_f32_16x16x32_bf16(false, a0, false, b0, (short)0, acc, false, false);
    acc = __builtin_amdgcn_wmma_f32_16x16x32_bf16(false, a1, false, b1, (short)0, acc, false, false);
    acc = __builtin_amdgcn_wmma_f32_16x16x32_bf16(false, a2, false, b2, (short)0, acc, false, false);
    acc = __builtin_amdgcn_wmma_f32_16x16x32_bf16(false, a3, false, b3, (short)0, acc, false, false);

    // C/D f32 16x16: VGPR r -> row r + 8*half, col = lane&15
    float* obase = out + (long long)(row0 + (half << 3)) * FOUT + n0 + nl;
    if (row0 + 16 <= nrows) {      // wave-uniform fast path: immediate-offset stores
#pragma unroll
        for (int r = 0; r < 8; ++r)
            obase[r * FOUT] = acc[r];
    } else {
#pragma unroll
        for (int r = 0; r < 8; ++r) {
            int row = row0 + (half << 3) + r;
            if (row < nrows) obase[r * FOUT] = acc[r];
        }
    }
}

// agg[dst,f..f+3] += h[src,f..f+3] * dis[src]*dis[dst]; one thread per (edge, 4 features)
__global__ void scatter_edges4(float* __restrict__ agg, const float* __restrict__ h,
                               const int* __restrict__ src, const int* __restrict__ dst,
                               const float* __restrict__ dis, long long total4, int logF) {
    long long gid = (long long)blockIdx.x * blockDim.x + threadIdx.x;
    if (gid >= total4) return;
    long long elem = gid << 2;
    int e = (int)(elem >> logF);
    int f = (int)(elem & ((1 << logF) - 1));
    int s = src[e], d = dst[e];               // wave-uniform for logF>=7
    float nrm = dis[s] * dis[d];
    float4 hv = *(const float4*)(h + ((long long)s << logF) + f);
    float* ap = agg + ((long long)d << logF) + f;
    atomicAdd(ap + 0, hv.x * nrm);
    atomicAdd(ap + 1, hv.y * nrm);
    atomicAdd(ap + 2, hv.z * nrm);
    atomicAdd(ap + 3, hv.w * nrm);
}

// out = (agg + h*dis[i]^2 + bias)[relu], float4-wide; out may alias agg.
__global__ void finalize_agg4(float* __restrict__ out, const float* __restrict__ agg,
                              const float* __restrict__ h, const float* __restrict__ dis,
                              const float* __restrict__ bias, long long total4, int logF,
                              int do_relu) {
    long long gid = (long long)blockIdx.x * blockDim.x + threadIdx.x;
    if (gid >= total4) return;
    long long elem = gid << 2;
    int i = (int)(elem >> logF);
    int f = (int)(elem & ((1 << logF) - 1));
    float di = dis[i];
    float sc = di * di;
    float4 av = *(const float4*)(agg + elem);
    float4 hv = *(const float4*)(h + elem);
    float4 bv = *(const float4*)(bias + f);
    float4 r;
    r.x = fmaf(hv.x, sc, av.x) + bv.x;
    r.y = fmaf(hv.y, sc, av.y) + bv.y;
    r.z = fmaf(hv.z, sc, av.z) + bv.z;
    r.w = fmaf(hv.w, sc, av.w) + bv.w;
    if (do_relu) {
        r.x = fmaxf(r.x, 0.f); r.y = fmaxf(r.y, 0.f);
        r.z = fmaxf(r.z, 0.f); r.w = fmaxf(r.w, 0.f);
    }
    *(float4*)(out + elem) = r;
}

extern "C" void kernel_launch(void* const* d_in, const int* in_sizes, int n_in,
                              void* d_out, int out_size, void* d_ws, size_t ws_size,
                              hipStream_t stream) {
    (void)n_in; (void)out_size; (void)ws_size;

    const float* x  = (const float*)d_in[0];
    const int*   ei = (const int*)d_in[1];
    const float* W1 = (const float*)d_in[2];
    const float* b1 = (const float*)d_in[3];
    const float* W2 = (const float*)d_in[4];
    const float* b2 = (const float*)d_in[5];
    const float* W3 = (const float*)d_in[6];
    const float* b3 = (const float*)d_in[7];
    float* out = (float*)d_out;

    const int N = GCN_N;
    const int E = in_sizes[1] / 2;
    const int* src = ei;
    const int* dst = ei + E;

    // workspace: dis[N] | buf0[N*128] | buf1[N*128] | Wt[128*128 bf16]  (~103 MB)
    float* dis  = (float*)d_ws;
    float* buf0 = dis + (((size_t)N + 255) & ~(size_t)255);
    float* buf1 = buf0 + (size_t)N * GCN_FIN;
    __bf16* Wt  = (__bf16*)(buf1 + (size_t)N * GCN_FIN);

    auto blks = [](long long n) { return (unsigned)((n + TB - 1) / TB); };
    const int rowTiles = (N + 15) / 16;

    // degree normalization
    zero_f32x4<<<blks(N / 4), TB, 0, stream>>>((float4*)dis, N / 4);
    deg_count <<<blks(E), TB, 0, stream>>>(dis, dst, E);
    deg_rsqrt <<<blks(N), TB, 0, stream>>>(dis, N);

    const long long NF4  = (long long)N * 128 / 4;
    const long long EF4  = (long long)E * 128 / 4;
    const long long NFo4 = (long long)N * 16 / 4;
    const long long EFo4 = (long long)E * 16 / 4;

    // ----- layer 1 -----
    w_bf16_transpose        <<<blks(128 * 128), TB, 0, stream>>>(Wt, W1, 128 * 128, 7);
    gemm_x128_bf16_wmma<128><<<rowTiles, TB, 0, stream>>>(buf1, x, Wt, N);
    zero_f32x4              <<<blks(NF4), TB, 0, stream>>>((float4*)buf0, NF4);
    scatter_edges4          <<<blks(EF4), TB, 0, stream>>>(buf0, buf1, src, dst, dis, EF4, 7);
    finalize_agg4           <<<blks(NF4), TB, 0, stream>>>(buf0, buf0, buf1, dis, b1, NF4, 7, 1);

    // ----- layer 2 -----
    w_bf16_transpose        <<<blks(128 * 128), TB, 0, stream>>>(Wt, W2, 128 * 128, 7);
    gemm_x128_bf16_wmma<128><<<rowTiles, TB, 0, stream>>>(buf1, buf0, Wt, N);
    zero_f32x4              <<<blks(NF4), TB, 0, stream>>>((float4*)buf0, NF4);
    scatter_edges4          <<<blks(EF4), TB, 0, stream>>>(buf0, buf1, src, dst, dis, EF4, 7);
    finalize_agg4           <<<blks(NF4), TB, 0, stream>>>(buf0, buf0, buf1, dis, b2, NF4, 7, 1);

    // ----- layer 3 (Fout=16) -----
    w_bf16_transpose        <<<blks(128 * 16), TB, 0, stream>>>(Wt, W3, 128 * 16, 4);
    gemm_x128_bf16_wmma<16> <<<rowTiles, TB, 0, stream>>>(buf1, buf0, Wt, N);
    zero_f32x4              <<<blks(NFo4), TB, 0, stream>>>((float4*)buf0, NFo4);
    scatter_edges4          <<<blks(EFo4), TB, 0, stream>>>(buf0, buf1, src, dst, dis, EFo4, 4);
    finalize_agg4           <<<blks(NFo4), TB, 0, stream>>>(out, buf0, buf1, dis, b3, NFo4, 4, 0);
}
